// GestureGRU_10496900072217
// MI455X (gfx1250) — matrix operand order repaired and checked by reading, not compiled, not run on hardware
//
#include <hip/hip_runtime.h>
#include <math.h>

// ---------------------------------------------------------------------------
// MI455X / gfx1250: wave32, WMMA 16x16x32 f16->f32.
// GRU: gates GEMMs are massively parallel (M=B*T=524288); the recurrent part
// is serial in T, so we keep w_hh (f16, ~102KB) and the f16 hidden state
// resident in LDS (320KB/WGP) and run a barrier-free per-wave scan where the
// r/z/n gate combination is perfectly lane-local in the WMMA C/D layout.
// Round 2: f16 hidden state in LDS (A-frags = contiguous ds_load_b128, no
// f32->f16 cvt staging) + __launch_bounds__(128,1) to kill scratch spills
// observed in the scan's inner loop.
// ---------------------------------------------------------------------------

typedef _Float16 f16;
typedef __attribute__((ext_vector_type(16))) _Float16 v16h;
typedef __attribute__((ext_vector_type(8)))  float    v8f;

#define T_STEPS 128
#define BATCH   4096
#define MTOT    (BATCH * T_STEPS)     // 524288 rows for the gates GEMMs

__device__ __forceinline__ v8f wmma_f16(v16h a, v16h b, v8f c) {
  // emits v_wmma_f32_16x16x32_f16
  return __builtin_amdgcn_wmma_f32_16x16x32_f16(false, a, false, b,
                                                (short)0, c, false, false);
}

// ---- fragment loaders (contiguous k order, same for A and B) ------------------
// A 16x32 f16: lane L in [0,16): row=L, K = {0..15} contiguous;
// lane L+16: same row, K = {16..31} (matches load_b_f16 k order).
__device__ __forceinline__ v16h load_a_f16(const f16* base, int ld, int lane, int kb) {
  int row = lane & 15;
  int kh  = (lane >> 4) << 4;            // 0 or 16
  const f16* p = base + row * ld + kb;
  v16h a;
#pragma unroll
  for (int i = 0; i < 8; ++i)  a[i] = p[kh + i];        // 16B contiguous
#pragma unroll
  for (int i = 8; i < 16; ++i) a[i] = p[kh + i];        // 16B contiguous
  return a;
}

// B 32x16 f16 from row-major W[N][ld] (B[k][n] = W[n][k], i.e. A*W^T):
// lane L: col = L&15, k = (L>>4)*16 + i, i=0..15 contiguous (32B).
__device__ __forceinline__ v16h load_b_f16(const f16* w, int ld, int lane, int nb, int kb) {
  int col = lane & 15;
  int kg  = (lane >> 4) << 4;            // 0 or 16
  const f16* p = w + (nb + col) * ld + kb + kg;
  v16h b;
#pragma unroll
  for (int i = 0; i < 16; ++i) b[i] = p[i];
  return b;
}

__device__ __forceinline__ float sigmoidf_(float x) {
  return 1.0f / (1.0f + __expf(-x));
}

// ---------------------------------------------------------------------------
// K0: pad/convert x f32 [M,63] -> f16 [M,64]
// ---------------------------------------------------------------------------
__global__ void xpad_kernel(const float* __restrict__ x, f16* __restrict__ xp, int total) {
  int i = blockIdx.x * blockDim.x + threadIdx.x;
  if (i >= total) return;
  int m = i >> 6, k = i & 63;
  xp[i] = (k < 63) ? (f16)x[(size_t)m * 63 + k] : (f16)0.0f;
}

// ---------------------------------------------------------------------------
// K1: gates = A(f16,[M,K]) @ W(f32,[384,Kw])^T + bias   -> G f32 [M,384]
// 256 threads = 8 waves, 128 M-rows per WG, W converted to f16 in LDS.
// ---------------------------------------------------------------------------
template <int K>
__global__ void gemm_gates_kernel(const f16* __restrict__ A, const float* __restrict__ W,
                                  int Kw, const float* __restrict__ bias,
                                  float* __restrict__ G) {
  constexpr int KP = K + 8;              // padded LDS stride (bank stagger)
  constexpr int KSTEPS = K / 32;
  extern __shared__ char smem[];
  f16* wlds = (f16*)smem;                // [384][KP]
  f16* alds = wlds + 384 * KP;           // [128][KP]

  int tid = threadIdx.x, wave = tid >> 5, lane = tid & 31;
  int c = lane & 15, rowb = (lane >> 4) << 3;

  for (int idx = tid; idx < 384 * K; idx += 256) {
    int n = idx / K, k = idx % K;
    wlds[n * KP + k] = (k < Kw) ? (f16)W[n * Kw + k] : (f16)0.0f;
  }
  const f16* Ablk = A + (size_t)blockIdx.x * 128 * K;
  for (int idx = tid; idx < 128 * K; idx += 256) {
    int r = idx / K, k = idx % K;
    alds[r * KP + k] = Ablk[idx];
  }
  __syncthreads();

  v16h af[KSTEPS];
#pragma unroll
  for (int kk = 0; kk < KSTEPS; ++kk)
    af[kk] = load_a_f16(alds + wave * 16 * KP, KP, lane, kk * 32);

  const v8f vzero = {};
  int mrow = blockIdx.x * 128 + wave * 16 + rowb;
  for (int nt = 0; nt < 24; ++nt) {
    v8f acc = vzero;
#pragma unroll
    for (int kk = 0; kk < KSTEPS; ++kk)
      acc = wmma_f16(af[kk], load_b_f16(wlds, KP, lane, nt * 16, kk * 32), acc);
    float bv = bias[nt * 16 + c];
    float* gp = G + (size_t)mrow * 384 + nt * 16 + c;
#pragma unroll
    for (int v = 0; v < 8; ++v) gp[(size_t)v * 384] = acc[v] + bv;
  }
}

// ---------------------------------------------------------------------------
// K2: recurrent scan. 128 batch rows per WG (4 waves x 32 rows, 2 M-tiles to
// halve per-row LDS reads of w_hh). gh = h @ w_hh^T via WMMA; gate math is
// lane-local: acc tile t -> r, t+8 -> z, t+16 -> n share (row,col) per
// lane/VGPR slot. Hidden state is f16 in LDS; T-loop is barrier-free (each
// wave reads/writes only its own 32 h rows; w_hh LDS is read-only).
// ---------------------------------------------------------------------------
#define WPAD 136   // f16 stride for w_hh  (128+8)
#define HPAD 136   // f16 stride for h     (128+8)

__global__ __launch_bounds__(128, 1)
void scan_kernel(const float* __restrict__ G, const float* __restrict__ whh,
                 const float* __restrict__ bhh, f16* __restrict__ h_all,
                 f16* __restrict__ h_last) {
  extern __shared__ char smem[];
  f16* wlds = (f16*)smem;                     // [384][WPAD]
  f16* hlds = (f16*)(smem + 384 * WPAD * 2);  // [128][HPAD]

  int tid = threadIdx.x, wave = tid >> 5, lane = tid & 31;
  int c = lane & 15, rowb = (lane >> 4) << 3;
  int bbase = blockIdx.x * 128;
  int rw0 = wave * 32;

  for (int idx = tid; idx < 384 * 128; idx += blockDim.x) {
    int n = idx >> 7, k = idx & 127;
    wlds[n * WPAD + k] = (f16)whh[idx];
  }
  for (int idx = tid; idx < 128 * HPAD; idx += blockDim.x) hlds[idx] = (f16)0.0f;

  float bhR[8], bhZ[8], bhN[8];
#pragma unroll
  for (int tt = 0; tt < 8; ++tt) {
    bhR[tt] = bhh[tt * 16 + c];
    bhZ[tt] = bhh[128 + tt * 16 + c];
    bhN[tt] = bhh[256 + tt * 16 + c];
  }
  __syncthreads();   // only barrier: after w_hh / h-state init

  const v8f vzero = {};
  for (int t = 0; t < T_STEPS; ++t) {
    v16h af[2][4];
#pragma unroll
    for (int rt = 0; rt < 2; ++rt)
#pragma unroll
      for (int kk = 0; kk < 4; ++kk)
        af[rt][kk] = load_a_f16(hlds + (rw0 + rt * 16) * HPAD, HPAD, lane, kk * 32);

#pragma unroll
    for (int tt = 0; tt < 8; ++tt) {
      v8f accR[2], accZ[2], accN[2];
#pragma unroll
      for (int rt = 0; rt < 2; ++rt) { accR[rt] = vzero; accZ[rt] = vzero; accN[rt] = vzero; }
#pragma unroll
      for (int kk = 0; kk < 4; ++kk) {
        v16h br = load_b_f16(wlds, WPAD, lane, tt * 16,        kk * 32);
        v16h bz = load_b_f16(wlds, WPAD, lane, (tt + 8) * 16,  kk * 32);
        v16h bn = load_b_f16(wlds, WPAD, lane, (tt + 16) * 16, kk * 32);
#pragma unroll
        for (int rt = 0; rt < 2; ++rt) {
          accR[rt] = wmma_f16(af[rt][kk], br, accR[rt]);
          accZ[rt] = wmma_f16(af[rt][kk], bz, accZ[rt]);
          accN[rt] = wmma_f16(af[rt][kk], bn, accN[rt]);
        }
      }
#pragma unroll
      for (int rt = 0; rt < 2; ++rt) {
        int rloc = rw0 + rt * 16 + rowb;
#pragma unroll
        for (int v = 0; v < 8; ++v) {
          int batch = bbase + rloc + v;
          int gbase = (batch * T_STEPS + t) * 384;
          float xr = G[gbase + tt * 16 + c];
          float xz = G[gbase + 128 + tt * 16 + c];
          float xn = G[gbase + 256 + tt * 16 + c];
          float r = sigmoidf_(xr + accR[rt][v] + bhR[tt]);
          float z = sigmoidf_(xz + accZ[rt][v] + bhZ[tt]);
          float n = tanhf(xn + r * (accN[rt][v] + bhN[tt]));
          float hp = (float)hlds[(rloc + v) * HPAD + tt * 16 + c];
          float hnew = (1.0f - z) * n + z * hp;
          hlds[(rloc + v) * HPAD + tt * 16 + c] = (f16)hnew;
          if (h_all) h_all[(batch * T_STEPS + t) * 128 + tt * 16 + c] = (f16)hnew;
          if (h_last && t == T_STEPS - 1) h_last[batch * 128 + tt * 16 + c] = (f16)hnew;
        }
      }
    }
  }
}

// ---------------------------------------------------------------------------
// K3: MLP head. hidden = relu(h2 @ w1^T + b1) [B,64]; out = hidden @ w2^T + b2.
// Two chained WMMA stages with f16 hidden staged in LDS; w2 padded 11->16.
// ---------------------------------------------------------------------------
__global__ void head_kernel(const f16* __restrict__ h2, const float* __restrict__ w1,
                            const float* __restrict__ b1, const float* __restrict__ w2,
                            const float* __restrict__ b2, float* __restrict__ out) {
  extern __shared__ char smem[];
  f16* w1lds = (f16*)smem;               // [64][136]
  f16* w2lds = w1lds + 64 * 136;         // [16][72]  (rows 11..15 zero)
  f16* hlds  = w2lds + 16 * 72;          // [128][136]
  f16* hid   = hlds + 128 * 136;         // [128][72]

  int tid = threadIdx.x, wave = tid >> 5, lane = tid & 31;
  int c = lane & 15, rowb = (lane >> 4) << 3;
  int mbase = blockIdx.x * 128;

  for (int idx = tid; idx < 64 * 128; idx += 256)
    w1lds[(idx >> 7) * 136 + (idx & 127)] = (f16)w1[idx];
  for (int idx = tid; idx < 16 * 72; idx += 256) {
    int r = idx / 72, k = idx % 72;
    w2lds[idx] = (r < 11 && k < 64) ? (f16)w2[r * 64 + k] : (f16)0.0f;
  }
  for (int idx = tid; idx < 128 * 128; idx += 256)
    hlds[(idx >> 7) * 136 + (idx & 127)] = h2[(size_t)mbase * 128 + idx];
  __syncthreads();

  const v8f vzero = {};
  v16h af[4];
#pragma unroll
  for (int kk = 0; kk < 4; ++kk)
    af[kk] = load_a_f16(hlds + wave * 16 * 136, 136, lane, kk * 32);

#pragma unroll
  for (int nt = 0; nt < 4; ++nt) {
    v8f acc = vzero;
#pragma unroll
    for (int kk = 0; kk < 4; ++kk)
      acc = wmma_f16(af[kk], load_b_f16(w1lds, 136, lane, nt * 16, kk * 32), acc);
    float bv = b1[nt * 16 + c];
#pragma unroll
    for (int v = 0; v < 8; ++v) {
      float hv = acc[v] + bv;
      hid[(wave * 16 + rowb + v) * 72 + nt * 16 + c] = (f16)fmaxf(hv, 0.0f);
    }
  }
  __syncthreads();

  v16h ag[2];
#pragma unroll
  for (int kk = 0; kk < 2; ++kk)
    ag[kk] = load_a_f16(hid + wave * 16 * 72, 72, lane, kk * 32);
  v8f acc = vzero;
#pragma unroll
  for (int kk = 0; kk < 2; ++kk)
    acc = wmma_f16(ag[kk], load_b_f16(w2lds, 72, lane, 0, kk * 32), acc);
  if (c < 11) {
    float bv = b2[c];
#pragma unroll
    for (int v = 0; v < 8; ++v)
      out[(size_t)(mbase + wave * 16 + rowb + v) * 11 + c] = acc[v] + bv;
  }
}

// ---------------------------------------------------------------------------
extern "C" void kernel_launch(void* const* d_in, const int* in_sizes, int n_in,
                              void* d_out, int out_size, void* d_ws, size_t ws_size,
                              hipStream_t stream) {
  (void)in_sizes; (void)n_in; (void)out_size; (void)ws_size;
  const float* x    = (const float*)d_in[0];
  const float* wih0 = (const float*)d_in[1];
  const float* whh0 = (const float*)d_in[2];
  const float* bih0 = (const float*)d_in[3];
  const float* bhh0 = (const float*)d_in[4];
  const float* wih1 = (const float*)d_in[5];
  const float* whh1 = (const float*)d_in[6];
  const float* bih1 = (const float*)d_in[7];
  const float* bhh1 = (const float*)d_in[8];
  const float* w1   = (const float*)d_in[9];
  const float* b1   = (const float*)d_in[10];
  const float* w2   = (const float*)d_in[11];
  const float* b2   = (const float*)d_in[12];
  float* out = (float*)d_out;

  // workspace layout (~1.01 GB), gates buffer reused by both layers
  char* ws = (char*)d_ws;
  f16*   xpad  = (f16*)ws;    ws += (size_t)MTOT * 64 * sizeof(f16);
  float* gates = (float*)ws;  ws += (size_t)MTOT * 384 * sizeof(float);
  f16*   h1    = (f16*)ws;    ws += (size_t)MTOT * 128 * sizeof(f16);
  f16*   h2l   = (f16*)ws;    ws += (size_t)BATCH * 128 * sizeof(f16);

  constexpr size_t smem_g64  = (size_t)(384 + 128) * 72  * sizeof(f16);  //  73728
  constexpr size_t smem_g128 = (size_t)(384 + 128) * 136 * sizeof(f16);  // 139264
  constexpr size_t smem_scan = (size_t)384 * WPAD * 2 + (size_t)128 * HPAD * 2; // 139264
  constexpr size_t smem_head = (size_t)(64 * 136 + 16 * 72 + 128 * 136 + 128 * 72) * sizeof(f16);

  (void)hipFuncSetAttribute((const void*)gemm_gates_kernel<64>,
        hipFuncAttributeMaxDynamicSharedMemorySize, (int)smem_g64);
  (void)hipFuncSetAttribute((const void*)gemm_gates_kernel<128>,
        hipFuncAttributeMaxDynamicSharedMemorySize, (int)smem_g128);
  (void)hipFuncSetAttribute((const void*)scan_kernel,
        hipFuncAttributeMaxDynamicSharedMemorySize, (int)smem_scan);
  (void)hipFuncSetAttribute((const void*)head_kernel,
        hipFuncAttributeMaxDynamicSharedMemorySize, (int)smem_head);

  const int xtot = MTOT * 64;
  xpad_kernel<<<(xtot + 255) / 256, 256, 0, stream>>>(x, xpad, xtot);

  // layer 0
  gemm_gates_kernel<64><<<MTOT / 128, 256, smem_g64, stream>>>(xpad, wih0, 63, bih0, gates);
  scan_kernel<<<BATCH / 128, 128, smem_scan, stream>>>(gates, whh0, bhh0, h1, (f16*)nullptr);
  // layer 1
  gemm_gates_kernel<128><<<MTOT / 128, 256, smem_g128, stream>>>(h1, wih1, 128, bih1, gates);
  scan_kernel<<<BATCH / 128, 128, smem_scan, stream>>>(gates, whh1, bhh1, (f16*)nullptr, h2l);
  // head
  head_kernel<<<BATCH / 128, 256, smem_head, stream>>>(h2l, w1, b1, w2, b2, out);
}
